// QValueDecoder_14757507629758
// MI455X (gfx1250) — matrix operand order repaired and checked by reading, not compiled, not run on hardware
//
#include <hip/hip_runtime.h>
#include <hip/hip_bf16.h>
#include <cstdint>

typedef _Float16 f16_t;
typedef __attribute__((ext_vector_type(16))) _Float16 v16h;
typedef __attribute__((ext_vector_type(8)))  _Float16 v8h;
typedef __attribute__((ext_vector_type(8)))  float    v8f;

#define N_AG   8
#define H_DIM  128
#define AT_DIM 128
#define NHEAD  4
#define E_DIM  32
#define TZ_DIM 64
#define NA_DIM 14
#define CATD   320
#define CATP   328   // padded f16 stride: 164 dwords -> rows land on distinct banks
#define QP     132   // padded f32 q stride
#define DTP    136   // padded f16 delta stride (17*16B)
#define BPW    8     // batches per workgroup
#define TPW    64    // tokens per workgroup (4 WMMA M-tiles of 16)

// ---------------- prep kernel: k/v tables + B-operand swizzles ----------------
__global__ __launch_bounds__(256)
void qvd_prep(const float* __restrict__ W_act,
              const float* __restrict__ Wq,
              const float* __restrict__ Wk, const float* __restrict__ bk,
              const float* __restrict__ Wv, const float* __restrict__ bv,
              const float* __restrict__ Wo,
              float* __restrict__ kt, float* __restrict__ vt,
              f16_t* __restrict__ Bq, f16_t* __restrict__ Bo)
{
    const int tid = threadIdx.x;
    // k_table[r][c] = relu(W_act[r])·Wk[:,c] + bk[c]; v_table with relu
    for (int t = tid; t < 15 * 128; t += 256) {
        int r = t >> 7, c = t & 127;
        float sk = bk[c], sv = bv[c];
        for (int h = 0; h < 128; h++) {
            float a = W_act[r * 128 + h];
            a = a > 0.f ? a : 0.f;
            sk += a * Wk[h * 128 + c];
            sv += a * Wv[h * 128 + c];
        }
        kt[t] = sk;
        vt[t] = sv > 0.f ? sv : 0.f;
    }
    // Wq -> B layout: [kstep(4)][ctile(8)][lane(32)][e(16)]
    // lane<16 holds K=ks*32+0..15, lane>=16 holds K=ks*32+16..31; col = ct*16 + lane%16
    for (int t = tid; t < 4 * 8 * 32 * 16; t += 256) {
        int e = t & 15, lane = (t >> 4) & 31, ct = (t >> 9) & 7, ks = t >> 12;
        int K = ks * 32 + ((lane & 16) ? 16 : 0) + e;
        int c = ct * 16 + (lane & 15);
        Bq[t] = (f16_t)Wq[K * 128 + c];
    }
    // Wo (320x14, pad cols to 16) -> B layout: [kstep(10)][lane(32)][e(16)]
    for (int t = tid; t < 10 * 32 * 16; t += 256) {
        int e = t & 15, lane = (t >> 4) & 31, ks = t >> 9;
        int K = ks * 32 + ((lane & 16) ? 16 : 0) + e;
        int c = lane & 15;
        Bo[t] = (c < NA_DIM) ? (f16_t)Wo[K * NA_DIM + c] : (f16_t)0.f;
    }
}

// ---------------- A-tile fetch from LDS into WMMA layout ----------------
__device__ __forceinline__ v16h load_a16(const f16_t* rowptr, int koff)
{
    // lane<16: K = base+0..15 contiguous; lane>=16: base+16..31 (koff includes +16)
    v8h lo = *(const v8h*)(rowptr + koff);
    v8h hi = *(const v8h*)(rowptr + koff + 8);
    v16h a;
#pragma unroll
    for (int t = 0; t < 8; t++) { a[t] = lo[t]; a[t + 8] = hi[t]; }
    return a;
}

// ---------------- main kernel ----------------
__global__ __launch_bounds__(128)
void qvd_main(const float* __restrict__ h_in, const float* __restrict__ onehot,
              const float* __restrict__ tz_in, const int* __restrict__ smask_in,
              const float* __restrict__ bq, const float* __restrict__ bo,
              const float* __restrict__ kv_g,          // k_table(1920 f32) ++ v_table(1920 f32)
              const v16h* __restrict__ Bq, const v16h* __restrict__ Bo,
              float* __restrict__ out, long long aux_off)
{
    __shared__ f16_t cat[TPW][CATP];   // [h(128) | extra(128) | tz(64)] per token, f16
    __shared__ float qld[TPW][QP];     // q = h@Wq + bq, f32
    __shared__ float kvtab[2 * 15 * 128]; // k_table then v_table, contiguous
    __shared__ f16_t dta[TPW][DTP];    // extra0 - extra_i, f16
    __shared__ int   sidx[TPW];        // shifted action index 0..14
    __shared__ int   smk[TPW];

    const int tid  = threadIdx.x;
    const int lane = tid & 31;
    const int wv   = tid >> 5;
    const int bb   = blockIdx.x;
    const long long g0 = (long long)bb * TPW;   // first global token

    // ---- phase 0a: k/v tables via CDNA5 async global->LDS DMA (no VGPR bounce) ----
    {
        const char* src = (const char*)kv_g;
        char* dst = (char*)kvtab;
        // 2*15*128 f32 = 15360 B = 960 x 16B chunks; per-wave iteration counts are
        // uniform, so EXEC is full at every async issue.
        for (int c = tid; c < 960; c += 128) {
            unsigned lds_off = (unsigned)(uintptr_t)(dst + c * 16);
            unsigned long long ga = (unsigned long long)(uintptr_t)(src + c * 16);
            asm volatile("global_load_async_to_lds_b128 %0, %1, off"
                         :: "v"(lds_off), "v"(ga) : "memory");
        }
    }

    // ---- phase 0b: stage inputs (with f32->f16 conversion) ----
    const float* hb = h_in + g0 * H_DIM;
    for (int t = tid; t < TPW * H_DIM; t += 128)
        cat[t >> 7][t & 127] = (f16_t)hb[t];
    const float* tzb = tz_in + g0 * TZ_DIM;
    for (int t = tid; t < TPW * TZ_DIM; t += 128)
        cat[t >> 6][256 + (t & 63)] = (f16_t)tzb[t];
    if (tid < TPW) {
        smk[tid] = smask_in[g0 + tid];
        int b_l = tid >> 3, j = tid & 7;
        int idx = 0;                                  // BOS row of W_act
        if (j > 0) {
            const float* row = onehot + ((long long)bb * BPW + b_l) * (N_AG * NA_DIM)
                               + (long long)(j - 1) * NA_DIM;
            for (int a2 = 0; a2 < NA_DIM; a2++)
                if (row[a2] > 0.5f) idx = a2 + 1;     // shifted by BOS slot
        }
        sidx[tid] = idx;
    }
    asm volatile("s_wait_asynccnt 0x0" ::: "memory"); // async DMA visible before barrier
    __syncthreads();

    const int mrow  = lane & 15;
    const int koff0 = (lane & 16) ? 16 : 0;
    const int rsel  = (lane & 16) ? 8 : 0;

    // ---- phase 1: Q = h @ Wq + bq via WMMA (16 tokens per wave) ----
    {
        const f16_t* arow = &cat[wv * 16 + mrow][0];
#pragma unroll
        for (int ct = 0; ct < 8; ct++) {
            float b0 = bq[ct * 16 + (lane & 15)];
            v8f acc = { b0, b0, b0, b0, b0, b0, b0, b0 };
#pragma unroll
            for (int ks = 0; ks < 4; ks++) {
                v16h a = load_a16(arow, ks * 32 + koff0);
                v16h b = Bq[(ks * 8 + ct) * 32 + lane];
                acc = __builtin_amdgcn_wmma_f32_16x16x32_f16(
                          false, a, false, b, (short)0, acc, false, false);
            }
#pragma unroll
            for (int r = 0; r < 8; r++)
                qld[wv * 16 + r + rsel][ct * 16 + (lane & 15)] = acc[r];
        }
    }
    __syncthreads();

    // ---- phase 2: masked causal attention (k/v are 15-entry tables) ----
    const float scale = 0.17677669529663687f;        // 1/sqrt(32)
    for (int task = tid; task < BPW * NHEAD * N_AG; task += 128) {
        int b_l = task >> 5, hh = (task >> 3) & 3, i = task & 7;
        int base = b_l * 8;
        int mi = smk[base + i];
        float qreg[E_DIM];
        const float* qrow = &qld[base + i][hh * 32];
#pragma unroll
        for (int e = 0; e < E_DIM; e++) qreg[e] = qrow[e];

        float s[8];
        float mx = -INFINITY;
#pragma unroll
        for (int j = 0; j < 8; j++) {
            s[j] = -INFINITY;
            if (j <= i && mi == 0 && smk[base + j] == 0) {
                const float* krow = &kvtab[sidx[base + j] * 128 + hh * 32];
                float d = 0.f;
#pragma unroll
                for (int e = 0; e < E_DIM; e++) d += qreg[e] * krow[e];
                s[j] = d * scale;
                mx = fmaxf(mx, s[j]);
            }
        }
        float acc[E_DIM];
#pragma unroll
        for (int e = 0; e < E_DIM; e++) acc[e] = 0.f;
        if (mx > -INFINITY) {
            float den = 0.f, w[8];
#pragma unroll
            for (int j = 0; j < 8; j++) {
                w[j] = (s[j] > -INFINITY) ? __expf(s[j] - mx) : 0.f;
                den += w[j];
            }
            float inv = 1.f / den;
#pragma unroll
            for (int j = 0; j < 8; j++) {
                float wj = w[j] * inv;
                if (wj != 0.f) {
                    const float* vrow = &kvtab[1920 + sidx[base + j] * 128 + hh * 32];
#pragma unroll
                    for (int e = 0; e < E_DIM; e++) acc[e] += wj * vrow[e];
                }
            }
        }
#pragma unroll
        for (int e = 0; e < E_DIM; e++)
            cat[base + i][128 + hh * 32 + e] = (f16_t)acc[e];
    }
    __syncthreads();

    // ---- phase 3: delta rows (extra0 - extra_i) for aux path ----
    for (int t = tid; t < TPW * 128; t += 128) {
        int tok = t >> 7, c = t & 127;
        int b0r = tok & ~7;
        dta[tok][c] = (f16_t)((float)cat[b0r][128 + c] - (float)cat[tok][128 + c]);
    }
    __syncthreads();

    // ---- phase 4: all_raw = cat@Wo (K=320), delta = dta@Wo[128:256] (K=128) ----
    {
        const f16_t* arow = &cat[wv * 16 + mrow][0];
        const f16_t* drow = &dta[wv * 16 + mrow][0];
        v8f acc  = { 0.f, 0.f, 0.f, 0.f, 0.f, 0.f, 0.f, 0.f };
        v8f acc2 = { 0.f, 0.f, 0.f, 0.f, 0.f, 0.f, 0.f, 0.f };
#pragma unroll
        for (int ks = 0; ks < 10; ks++) {
            v16h a = load_a16(arow, ks * 32 + koff0);
            v16h b = Bo[ks * 32 + lane];
            acc = __builtin_amdgcn_wmma_f32_16x16x32_f16(
                      false, a, false, b, (short)0, acc, false, false);
        }
#pragma unroll
        for (int ks = 0; ks < 4; ks++) {
            v16h a = load_a16(drow, ks * 32 + koff0);
            v16h b = Bo[(4 + ks) * 32 + lane];     // Wo rows 128..255 = extra slice
            acc2 = __builtin_amdgcn_wmma_f32_16x16x32_f16(
                       false, a, false, b, (short)0, acc2, false, false);
        }
        int a_col = lane & 15;
        if (a_col < NA_DIM) {
            float boa = bo[a_col];
#pragma unroll
            for (int r = 0; r < 8; r++) {
                int tl = wv * 16 + r + rsel;
                long long gt = g0 + tl;
                float raw  = acc[r] + boa;
                float allv = (((tl & 7) == 0) || (smk[tl] != 0)) ? 0.f : raw;
                out[gt * NA_DIM + a_col] = allv;
                out[aux_off + gt * NA_DIM + a_col] = raw + acc2[r];
            }
        }
    }
}

// ---------------- launch ----------------
extern "C" void kernel_launch(void* const* d_in, const int* in_sizes, int n_in,
                              void* d_out, int out_size, void* d_ws, size_t ws_size,
                              hipStream_t stream)
{
    const float* h_in   = (const float*)d_in[0];
    const float* onehot = (const float*)d_in[1];
    const float* tz     = (const float*)d_in[2];
    const int*   sm     = (const int*)d_in[3];
    const float* W_act  = (const float*)d_in[4];
    const float* Wq     = (const float*)d_in[5];
    const float* bq     = (const float*)d_in[6];
    const float* Wk     = (const float*)d_in[7];
    const float* bk     = (const float*)d_in[8];
    const float* Wv     = (const float*)d_in[9];
    const float* bv     = (const float*)d_in[10];
    const float* Wo     = (const float*)d_in[11];
    const float* bo     = (const float*)d_in[12];
    float* out = (float*)d_out;

    char* ws = (char*)d_ws;
    float* kt    = (float*)ws;              //  7680 B: k_table[15][128]
    float* vt    = (float*)(ws + 7680);     //  7680 B: v_table[15][128] (contiguous after kt)
    f16_t* Bq_sw = (f16_t*)(ws + 15360);    // 32768 B: swizzled Wq
    f16_t* Bo_sw = (f16_t*)(ws + 48128);    // 10240 B: swizzled Wo

    int BS = in_sizes[0] / (N_AG * H_DIM);  // 32768

    qvd_prep<<<1, 256, 0, stream>>>(W_act, Wq, Wk, bk, Wv, bv, Wo, kt, vt, Bq_sw, Bo_sw);

    long long aux = (long long)out_size / 2;
    qvd_main<<<BS / BPW, 128, 0, stream>>>(h_in, onehot, tz, sm, bq, bo, kt,
                                           (const v16h*)Bq_sw, (const v16h*)Bo_sw,
                                           out, aux);
}
